// DecoderOnlyTransformer_10041633538673
// MI455X (gfx1250) — matrix operand order repaired and checked by reading, not compiled, not run on hardware
//
#include <hip/hip_runtime.h>
#include <hip/hip_bf16.h>
#include <math.h>

typedef __attribute__((ext_vector_type(16))) _Float16 v16h;
typedef __attribute__((ext_vector_type(8)))  float    v8f;

union Frag16 { v16h v; uint4 u[2]; };

// ---------------------------------------------------------------------------
// Fragment layouts per CDNA5 ISA 7.12.2 (wave32), f16 16x16x32:
//   A 16x32 : lane l, m=l&15, halves: k = (i<8?0:16) + (l>>4)*8 + (i&7)
//   B 32x16 : lane l, n=l&15, k = (i<8?0:16) + (l>>4)*8 + (i&7)   (B supplied transposed [N,K])
//   C 16x16 : lane l, n=l&15, m = r + 8*(l>>4)
// ---------------------------------------------------------------------------

// ---------------------------------------------------------------------------
// LDS-staged WMMA GEMM (CDNA5 async path):
//   C[M,N] = scale * A[M,K] * Bt[N,K]^T,  M,N multiples of 128, K mult of 32.
// 256 threads = 8 waves (4 M x 2 N), wave tile 32x64 (2x4 WMMA tiles),
// workgroup tile 128x128. Per K-step the 128x32 A and B tiles are staged into
// double-buffered LDS with GLOBAL_LOAD_ASYNC_TO_LDS_B128 (ASYNCcnt), then
// consumed via ds_load_b128. Epilogue: optional f32 / f16 / transposed-f16.
// ---------------------------------------------------------------------------
__global__ __launch_bounds__(256)
void wmma_gemm_lds(const _Float16* __restrict__ A, int lda,
                   const _Float16* __restrict__ Bt, int ldb,
                   float* __restrict__ C, int ldc,
                   _Float16* __restrict__ Ch, int ldch,
                   _Float16* __restrict__ Ct, int ldct,
                   int Kdim, float scale)
{
    __shared__ __align__(16) _Float16 As[2][128][32];
    __shared__ __align__(16) _Float16 Bs[2][128][32];

    const int tid  = threadIdx.x;
    const int wave = tid >> 5;
    const int lane = tid & 31;
    const int wm   = wave >> 1;        // 0..3 -> 32-row slice
    const int wn   = wave & 1;         // 0..1 -> 64-col slice
    const int half = lane >> 4;
    const int l15  = lane & 15;
    const int m0   = blockIdx.y * 128;
    const int n0   = blockIdx.x * 128;

    v8f acc[2][4] = {};

    // Stage one 128x32 f16 tile of A and of Bt into LDS buffer `buf`.
    // 512 16-byte chunks per tile; 2 per thread per matrix.
    auto stage = [&](int buf, int k0) {
#pragma unroll
        for (int c = 0; c < 2; ++c) {
            const int idx = tid + c * 256;
            const int row = idx >> 2;
            const int ch  = idx & 3;
            const _Float16* ga = A  + (size_t)(m0 + row) * lda + k0 + ch * 8;
            const _Float16* gb = Bt + (size_t)(n0 + row) * ldb + k0 + ch * 8;
            const uint32_t la = (uint32_t)(uintptr_t)(&As[buf][row][ch * 8]);
            const uint32_t lb = (uint32_t)(uintptr_t)(&Bs[buf][row][ch * 8]);
            asm volatile("global_load_async_to_lds_b128 %0, %1, off"
                         :: "v"(la), "v"(ga) : "memory");
            asm volatile("global_load_async_to_lds_b128 %0, %1, off"
                         :: "v"(lb), "v"(gb) : "memory");
        }
    };

    stage(0, 0);
    asm volatile("s_wait_asynccnt 0x0" ::: "memory");
    __syncthreads();

    int cur = 0;
    for (int k0 = 0; k0 < Kdim; k0 += 32) {
        if (k0 + 32 < Kdim) stage(cur ^ 1, k0 + 32);   // async prefetch next

        Frag16 a[2], b[4];
#pragma unroll
        for (int i = 0; i < 2; ++i) {
            const _Float16* ar = &As[cur][wm * 32 + i * 16 + l15][0];
            a[i].u[0] = *(const uint4*)(ar + half * 8);
            a[i].u[1] = *(const uint4*)(ar + 16 + half * 8);
        }
#pragma unroll
        for (int j = 0; j < 4; ++j) {
            const _Float16* br = &Bs[cur][wn * 64 + j * 16 + l15][half * 8];
            b[j].u[0] = *(const uint4*)(br);
            b[j].u[1] = *(const uint4*)(br + 16);
        }
#pragma unroll
        for (int i = 0; i < 2; ++i)
#pragma unroll
            for (int j = 0; j < 4; ++j)
                acc[i][j] = __builtin_amdgcn_wmma_f32_16x16x32_f16(
                    false, a[i].v, false, b[j].v, (short)0, acc[i][j], false, false);

        asm volatile("s_wait_asynccnt 0x0" ::: "memory");
        __syncthreads();
        cur ^= 1;
    }

#pragma unroll
    for (int i = 0; i < 2; ++i)
#pragma unroll
        for (int j = 0; j < 4; ++j) {
            const int mt = m0 + wm * 32 + i * 16;
            const int n  = n0 + wn * 64 + j * 16 + l15;
            if (C) {
#pragma unroll
                for (int r = 0; r < 8; ++r)
                    C[(size_t)(mt + r + 8 * half) * ldc + n] = acc[i][j][r] * scale;
            }
            if (Ch) {
#pragma unroll
                for (int r = 0; r < 8; ++r)
                    Ch[(size_t)(mt + r + 8 * half) * ldch + n] =
                        (_Float16)(acc[i][j][r] * scale);
            }
            if (Ct) {
                union { uint4 u; _Float16 h[8]; } pk;
#pragma unroll
                for (int r = 0; r < 8; ++r) pk.h[r] = (_Float16)(acc[i][j][r] * scale);
                *(uint4*)(Ct + (size_t)n * ldct + mt + 8 * half) = pk.u;
            }
        }
}

// ---------------------------------------------------------------------------
// Direct-from-global WMMA GEMM (kept for N=64 attention-output GEMM).
// 128 threads = 4 waves (2x2), wave tile 32x32, wg tile 64x64.
// ---------------------------------------------------------------------------
__global__ __launch_bounds__(128)
void wmma_gemm(const _Float16* __restrict__ A, int lda,
               const _Float16* __restrict__ Bt, int ldb,
               float* __restrict__ C, int ldc,
               _Float16* __restrict__ Ch, int ldch,
               _Float16* __restrict__ Ct, int ldct,
               int M, int N, int Kdim, float scale)
{
    const int wave = threadIdx.x >> 5;
    const int lane = threadIdx.x & 31;
    const int wm = wave >> 1, wn = wave & 1;
    const int m0 = blockIdx.y * 64 + wm * 32;
    const int n0 = blockIdx.x * 64 + wn * 32;
    const int half = lane >> 4;
    const int l15  = lane & 15;

    v8f acc[2][2] = {};

    for (int k0 = 0; k0 < Kdim; k0 += 32) {
        Frag16 a[2], b[2];
#pragma unroll
        for (int i = 0; i < 2; ++i) {
            const _Float16* ar = A + (size_t)(m0 + i * 16 + l15) * lda + k0;
            a[i].u[0] = *(const uint4*)(ar + half * 8);
            a[i].u[1] = *(const uint4*)(ar + 16 + half * 8);
            const _Float16* br = Bt + (size_t)(n0 + i * 16 + l15) * ldb + k0 + half * 8;
            b[i].u[0] = *(const uint4*)(br);
            b[i].u[1] = *(const uint4*)(br + 16);
        }
#pragma unroll
        for (int i = 0; i < 2; ++i)
#pragma unroll
            for (int j = 0; j < 2; ++j)
                acc[i][j] = __builtin_amdgcn_wmma_f32_16x16x32_f16(
                    false, a[i].v, false, b[j].v, (short)0, acc[i][j], false, false);
    }

#pragma unroll
    for (int i = 0; i < 2; ++i)
#pragma unroll
        for (int j = 0; j < 2; ++j) {
            const int mt = m0 + i * 16;
            const int n  = n0 + j * 16 + l15;
            if (C) {
#pragma unroll
                for (int r = 0; r < 8; ++r)
                    C[(size_t)(mt + r + 8 * half) * ldc + n] = acc[i][j][r] * scale;
            }
            if (Ch) {
#pragma unroll
                for (int r = 0; r < 8; ++r)
                    Ch[(size_t)(mt + r + 8 * half) * ldch + n] =
                        (_Float16)(acc[i][j][r] * scale);
            }
            if (Ct) {
                union { uint4 u; _Float16 h[8]; } pk;
#pragma unroll
                for (int r = 0; r < 8; ++r) pk.h[r] = (_Float16)(acc[i][j][r] * scale);
                *(uint4*)(Ct + (size_t)n * ldct + mt + 8 * half) = pk.u;
            }
        }
}

// f32 -> f16 elementwise
__global__ void cvt_f16(const float* __restrict__ X, _Float16* __restrict__ Y, int n)
{
    int i = blockIdx.x * blockDim.x + threadIdx.x;
    if (i < n) Y[i] = (_Float16)X[i];
}

// f32 [rows,cols] -> f16 transposed [cols,rows], LDS-tiled
__global__ void transpose_to_f16(const float* __restrict__ W, int rows, int cols,
                                 _Float16* __restrict__ WT)
{
    __shared__ float tile[32][33];
    const int r0 = blockIdx.y * 32, c0 = blockIdx.x * 32;
    const int tx = threadIdx.x, ty = threadIdx.y;
    for (int i = ty; i < 32; i += 8)
        tile[i][tx] = W[(size_t)(r0 + i) * cols + c0 + tx];
    __syncthreads();
    for (int i = ty; i < 32; i += 8)
        WT[(size_t)(c0 + i) * rows + r0 + tx] = (_Float16)tile[tx][i];
}

// per-row max and sum(exp(x-max)) for softmax over the row (last axis)
__global__ void row_softmax_stats(const float* __restrict__ S, int ld, int ncol,
                                  float* __restrict__ mOut, float* __restrict__ zOut)
{
    __shared__ float red[256];
    const int row = blockIdx.x;
    const float* r = S + (size_t)row * ld;
    float mx = -INFINITY;
    for (int c = threadIdx.x; c < ncol; c += 256) mx = fmaxf(mx, r[c]);
    red[threadIdx.x] = mx; __syncthreads();
    for (int s = 128; s > 0; s >>= 1) {
        if (threadIdx.x < s) red[threadIdx.x] = fmaxf(red[threadIdx.x], red[threadIdx.x + s]);
        __syncthreads();
    }
    mx = red[0]; __syncthreads();
    float sum = 0.f;
    for (int c = threadIdx.x; c < ncol; c += 256) sum += expf(r[c] - mx);
    red[threadIdx.x] = sum; __syncthreads();
    for (int s = 128; s > 0; s >>= 1) {
        if (threadIdx.x < s) red[threadIdx.x] += red[threadIdx.x + s];
        __syncthreads();
    }
    if (threadIdx.x == 0) { mOut[row] = mx; zOut[row] = red[0]; }
}

// PT[q,k] = exp(S[k,q]-m[k])/z[k] as f16, via LDS tile transpose
__global__ void exp_transpose(const float* __restrict__ S, int ld,
                              const float* __restrict__ m, const float* __restrict__ z,
                              _Float16* __restrict__ PT, int ldt)
{
    __shared__ float tile[32][33];
    const int k0 = blockIdx.y * 32, q0 = blockIdx.x * 32;
    const int tx = threadIdx.x, ty = threadIdx.y;
    for (int i = ty; i < 32; i += 8) {
        const int k = k0 + i;
        tile[i][tx] = expf(S[(size_t)k * ld + q0 + tx] - m[k]) / z[k];
    }
    __syncthreads();
    for (int i = ty; i < 32; i += 8)
        PT[(size_t)(q0 + i) * ldt + k0 + tx] = (_Float16)tile[tx][i];
}

// row L2-normalize (eps=1e-12), f16 output
__global__ void l2norm_f16(const float* __restrict__ X, _Float16* __restrict__ Y, int ncol)
{
    __shared__ float red[256];
    const int row = blockIdx.x;
    const float* r = X + (size_t)row * ncol;
    float s = 0.f;
    for (int c = threadIdx.x; c < ncol; c += 256) { float v = r[c]; s += v * v; }
    red[threadIdx.x] = s; __syncthreads();
    for (int t = 128; t > 0; t >>= 1) {
        if (threadIdx.x < t) red[threadIdx.x] += red[threadIdx.x + t];
        __syncthreads();
    }
    const float inv = 1.0f / fmaxf(sqrtf(red[0]), 1e-12f);
    for (int c = threadIdx.x; c < ncol; c += 256)
        Y[(size_t)row * ncol + c] = (_Float16)(r[c] * inv);
}

// row L2-normalize + exact-erf GELU, f32 output
__global__ void l2norm_gelu(const float* __restrict__ X, float* __restrict__ Y, int ncol)
{
    __shared__ float red[256];
    const int row = blockIdx.x;
    const float* r = X + (size_t)row * ncol;
    float s = 0.f;
    for (int c = threadIdx.x; c < ncol; c += 256) { float v = r[c]; s += v * v; }
    red[threadIdx.x] = s; __syncthreads();
    for (int t = 128; t > 0; t >>= 1) {
        if (threadIdx.x < t) red[threadIdx.x] += red[threadIdx.x + t];
        __syncthreads();
    }
    const float inv = 1.0f / fmaxf(sqrtf(red[0]), 1e-12f);
    for (int c = threadIdx.x; c < ncol; c += 256) {
        float v = r[c] * inv;
        Y[(size_t)row * ncol + c] = v * 0.5f * (1.0f + erff(v * 0.70710678118654752f));
    }
}

extern "C" void kernel_launch(void* const* d_in, const int* in_sizes, int n_in,
                              void* d_out, int out_size, void* d_ws, size_t ws_size,
                              hipStream_t stream)
{
    (void)in_sizes; (void)n_in; (void)out_size; (void)ws_size;
    const float* x   = (const float*)d_in[0];
    const float* Wq  = (const float*)d_in[1];
    const float* Wk  = (const float*)d_in[2];
    const float* Wv  = (const float*)d_in[3];
    const float* Wo  = (const float*)d_in[4];
    const float* Wff = (const float*)d_in[5];
    float* out = (float*)d_out;

    const int B = 2, S = 2048, D = 1024, L = 1024, H = 16, dh = 64;
    const int MX = B * S;                 // 4096
    const float inv_sqrtD = 0.03125f;     // 1/sqrt(1024)

    // Workspace layout (90 MiB total); attention scratch aliases post-attn bufs.
    char* ws = (char*)d_ws;
    const size_t Mi = 1ull << 20;
    _Float16* xh   = (_Float16*)(ws +  0 * Mi);   //  8 MiB  x f16 [4096,1024]
    _Float16* WqT  = (_Float16*)(ws +  8 * Mi);   //  2 MiB  Wq^T f16 [1024,1024]
    _Float16* WkT  = (_Float16*)(ws + 10 * Mi);
    _Float16* WvT  = (_Float16*)(ws + 12 * Mi);
    _Float16* WoT  = (_Float16*)(ws + 14 * Mi);
    _Float16* WffT = (_Float16*)(ws + 16 * Mi);
    _Float16* Qh   = (_Float16*)(ws + 18 * Mi);   //  8 MiB  Q f16 [4096,1024]
    _Float16* Kh   = (_Float16*)(ws + 26 * Mi);   //  8 MiB
    _Float16* VT   = (_Float16*)(ws + 34 * Mi);   //  8 MiB  V^T f16 [1024,4096]
    _Float16* z1   = (_Float16*)(ws + 42 * Mi);   //  8 MiB  attn out f16 [4096,1024]
    float*    Sc   = (float*)   (ws + 50 * Mi);   // 16 MiB  scores f32 [2048,2048]
    _Float16* PT   = (_Float16*)(ws + 66 * Mi);   //  8 MiB  attn^T f16 [2048,2048]
    float*    mZ   = (float*)   (ws + 74 * Mi);   // 16 KiB  row max / sumexp
    float*    z2   = (float*)   (ws + 50 * Mi);   // 16 MiB  f32 (after attention)
    _Float16* z3   = (_Float16*)(ws + 66 * Mi);   //  8 MiB  f16
    float*    z4   = (float*)   (ws + 74 * Mi);   // 16 MiB  f32

    const dim3 tblk(32, 8), tgrd(32, 32);

    // 1) Precision staging: x -> f16; weights -> transposed f16.
    cvt_f16<<<(MX * D + 255) / 256, 256, 0, stream>>>(x, xh, MX * D);
    transpose_to_f16<<<tgrd, tblk, 0, stream>>>(Wq,  D, L, WqT);
    transpose_to_f16<<<tgrd, tblk, 0, stream>>>(Wk,  D, L, WkT);
    transpose_to_f16<<<tgrd, tblk, 0, stream>>>(Wv,  D, L, WvT);
    transpose_to_f16<<<tgrd, tblk, 0, stream>>>(Wo,  L, D, WoT);
    transpose_to_f16<<<tgrd, tblk, 0, stream>>>(Wff, D, D, WffT);

    // 2) Q/K/V projections via async-LDS WMMA GEMM (V written transposed).
    const dim3 gProj(L / 128, MX / 128);
    wmma_gemm_lds<<<gProj, 256, 0, stream>>>(xh, D, WqT, D, nullptr, 0, Qh, L,
                                             nullptr, 0, D, 1.0f);
    wmma_gemm_lds<<<gProj, 256, 0, stream>>>(xh, D, WkT, D, nullptr, 0, Kh, L,
                                             nullptr, 0, D, 1.0f);
    wmma_gemm_lds<<<gProj, 256, 0, stream>>>(xh, D, WvT, D, nullptr, 0, nullptr, 0,
                                             VT, MX, D, 1.0f);

    // 3) Attention per (batch, head): scores = K·Q^T/32 -> softmax over q
    //    (reference quirk: softmax over the LAST axis of [k,q]) -> out = attn^T·V.
    for (int b = 0; b < B; ++b)
        for (int h = 0; h < H; ++h) {
            const _Float16* Khd = Kh + (size_t)b * S * L + h * dh; // A [S,dh], lda=L
            const _Float16* Qhd = Qh + (size_t)b * S * L + h * dh; // Bt [S,dh], ldb=L
            wmma_gemm_lds<<<dim3(S / 128, S / 128), 256, 0, stream>>>(
                Khd, L, Qhd, L, Sc, S, nullptr, 0, nullptr, 0, dh, inv_sqrtD);

            row_softmax_stats<<<S, 256, 0, stream>>>(Sc, S, S, mZ, mZ + S);
            exp_transpose<<<dim3(S / 32, S / 32), tblk, 0, stream>>>(Sc, S, mZ, mZ + S,
                                                                     PT, S);

            const _Float16* Vhd = VT + (size_t)h * dh * MX + (size_t)b * S; // Bt [dh,S]
            _Float16* z1hd = z1 + (size_t)b * S * L + h * dh;
            wmma_gemm<<<dim3(dh / 64, S / 64), 128, 0, stream>>>(
                PT, S, Vhd, MX, nullptr, 0, z1hd, L, nullptr, 0, S, dh, S, 1.0f);
        }

    // 4) Output projection, normalize, FFN, normalize + exact GELU.
    wmma_gemm_lds<<<dim3(D / 128, MX / 128), 256, 0, stream>>>(
        z1, L, WoT, L, z2, D, nullptr, 0, nullptr, 0, L, 1.0f);
    l2norm_f16<<<MX, 256, 0, stream>>>(z2, z3, D);
    wmma_gemm_lds<<<dim3(D / 128, MX / 128), 256, 0, stream>>>(
        z3, D, WffT, D, z4, D, nullptr, 0, nullptr, 0, D, 1.0f);
    l2norm_gelu<<<MX, 256, 0, stream>>>(z4, out, D);
}
